// SnailAttention_32409823216081
// MI455X (gfx1250) — matrix operand order repaired and checked by reading, compiled, not run on hardware
//
#include <hip/hip_runtime.h>

typedef __bf16 bf16_t;
typedef bf16_t v16bf __attribute__((ext_vector_type(16)));
typedef bf16_t v8bf  __attribute__((ext_vector_type(8)));
typedef float  v8f   __attribute__((ext_vector_type(8)));
typedef float  v4f   __attribute__((ext_vector_type(4)));

constexpr int BATCH = 4;
constexpr int SEQ   = 4096;
constexpr int DIM   = 128;
constexpr int HD    = 64;        // head dim for q/k and v
constexpr int OUTW  = DIM + HD;  // 192
// softmax scale 1/sqrt(64) and log2(e) folded into Q at projection time,
// so the flash inner loop uses raw exp2 on (s - m).
constexpr float QSCALE  = 0.125f * 1.4426950408889634f;
constexpr float NEG_BIG = -3.0e38f;

// ---------------------------------------------------------------- helpers

__device__ inline v8f vzero8() {
  v8f r;
#pragma unroll
  for (int i = 0; i < 8; ++i) r[i] = 0.0f;
  return r;
}

__device__ inline v8f wmma_bf16(v16bf a, v16bf b, v8f c) {
  // D = A(16x32 bf16) x B(32x16 bf16) + C(16x16 f32)
  return __builtin_amdgcn_wmma_f32_16x16x32_bf16(
      /*neg_a=*/false, a, /*neg_b=*/false, b,
      /*c_mod=*/(short)0, c, /*reuse_a=*/false, /*reuse_b=*/false);
}

// A-matrix fragment (16-bit, 16x32): lane holds row M=lane&15,
// K elements {k0..k0+15} with k0 = kbase + 16*(lane>=16).
__device__ inline v16bf load_a_bf16(const bf16_t* rowbase, int k0) {
  v8bf lo = *(const v8bf*)(rowbase + k0);
  v8bf hi = *(const v8bf*)(rowbase + k0 + 8);
  v16bf r;
#pragma unroll
  for (int i = 0; i < 8; ++i) { r[i] = lo[i]; r[i + 8] = hi[i]; }
  return r;
}

__device__ inline v16bf load_a_f32(const float* rowbase, int k0) {
  v4f a0 = *(const v4f*)(rowbase + k0);
  v4f a1 = *(const v4f*)(rowbase + k0 + 4);
  v4f b0 = *(const v4f*)(rowbase + k0 + 8);
  v4f b1 = *(const v4f*)(rowbase + k0 + 12);
  v16bf r;
#pragma unroll
  for (int i = 0; i < 4; ++i) {
    r[i]      = (bf16_t)a0[i];
    r[i + 4]  = (bf16_t)a1[i];
    r[i + 8]  = (bf16_t)b0[i];
    r[i + 12] = (bf16_t)b1[i];
  }
  return r;
}

// ---------------------------------------------------------------- kernel 0:
// W[128][64] f32 -> Wt[64][128] bf16 (x3 matrices packed consecutively)
__global__ void prep_weights_kernel(const float* __restrict__ Wq,
                                    const float* __restrict__ Wk,
                                    const float* __restrict__ Wv,
                                    bf16_t* __restrict__ Wt) {
  int i = blockIdx.x * blockDim.x + threadIdx.x;   // 3*64*128 total
  if (i >= 3 * HD * DIM) return;
  int m   = i / (HD * DIM);
  int rem = i - m * (HD * DIM);
  int col = rem / DIM;
  int k   = rem - col * DIM;
  const float* W = (m == 0) ? Wq : ((m == 1) ? Wk : Wv);
  Wt[i] = (bf16_t)W[k * HD + col];
}

// ---------------------------------------------------------------- kernel 1:
// QKV projection with WMMA. 4 waves/block, wave w owns output cols [16w,16w+16)
// for a 16-row tile. Writes Qb (pre-scaled) / Kb row-major bf16, Vt transposed.
__global__ __launch_bounds__(128) void qkv_proj_kernel(
    const float* __restrict__ x, const bf16_t* __restrict__ Wt,
    const float* __restrict__ bq, const float* __restrict__ bk,
    const float* __restrict__ bv, bf16_t* __restrict__ Qb,
    bf16_t* __restrict__ Kb, bf16_t* __restrict__ Vt) {
  const int wave  = threadIdx.x >> 5;
  const int lane  = threadIdx.x & 31;
  const int l15   = lane & 15;
  const int h     = lane >> 4;
  const int khalf = h * 16;    // A-fragment half offset
  const int klo   = h * 16;    // B-fragment half offset
  const int rbase = blockIdx.x * 16;
  const int colLane = wave * 16 + l15;

  const float*  xrow = x + (size_t)(rbase + l15) * DIM;
  const bf16_t* WtQ  = Wt + (size_t)colLane * DIM;
  const bf16_t* WtK  = WtQ + HD * DIM;
  const bf16_t* WtV  = WtK + HD * DIM;

  v8f aq = vzero8(), ak = vzero8(), av = vzero8();
#pragma unroll
  for (int ks = 0; ks < 4; ++ks) {               // K = 128 in 4 steps of 32
    v16bf a  = load_a_f32(xrow, ks * 32 + khalf);
    v16bf wq = *(const v16bf*)(WtQ + ks * 32 + klo);
    v16bf wk = *(const v16bf*)(WtK + ks * 32 + klo);
    v16bf wv = *(const v16bf*)(WtV + ks * 32 + klo);
    aq = wmma_bf16(a, wq, aq);
    ak = wmma_bf16(a, wk, ak);
    av = wmma_bf16(a, wv, av);
  }
  const float biasq = bq[colLane], biask = bk[colLane], biasv = bv[colLane];
#pragma unroll
  for (int r = 0; r < 8; ++r) {
    const int row = rbase + r + h * 8;           // global (b*S + s) row
    Qb[(size_t)row * HD + colLane] = (bf16_t)((aq[r] + biasq) * QSCALE);
    Kb[(size_t)row * HD + colLane] = (bf16_t)(ak[r] + biask);
    const int b = row >> 12;                     // SEQ == 4096
    const int s = row & (SEQ - 1);
    Vt[((size_t)b * HD + colLane) * SEQ + s] = (bf16_t)(av[r] + biasv);
  }
}

// ---------------------------------------------------------------- kernel 2:
// out[:, :128] = inputs  (row stride 192 floats)
__global__ void copy_inputs_kernel(const float4* __restrict__ x4,
                                   float* __restrict__ out) {
  int t = blockIdx.x * blockDim.x + threadIdx.x;  // B*S*32 total
  int row = t >> 5, j = t & 31;
  float4 v = x4[t];
  *((float4*)(out + (size_t)row * OUTW) + j) = v;
}

// ---------------------------------------------------------------- kernel 3:
// Fused causal flash attention. 4 waves/block; wave owns 16 queries.
// Key blocks of 64: all 4 waves need exactly blockIdx.x+1 blocks, and only
// the last block is diagonal (interior blocks need no masking at all).
__global__ __launch_bounds__(128) void attn_kernel(
    const bf16_t* __restrict__ Qb, const bf16_t* __restrict__ Kb,
    const bf16_t* __restrict__ Vt, float* __restrict__ out) {
  __shared__ bf16_t psh[4 * 16 * 64];   // per-wave 16x64 P tile (row-major)

  const int wave  = threadIdx.x >> 5;
  const int lane  = threadIdx.x & 31;
  const int l15   = lane & 15;
  const int h     = lane >> 4;
  const int khalf = h * 16;
  const int klo   = h * 16;
  const int b     = blockIdx.y;
  const int q0    = blockIdx.x * 64 + wave * 16;

  const bf16_t* Qrow  = Qb + (size_t)(b * SEQ + q0 + l15) * HD;
  const bf16_t* Kbase = Kb + (size_t)b * SEQ * HD;
  const bf16_t* Vtb   = Vt + (size_t)b * HD * SEQ;
  bf16_t* pw = psh + wave * (16 * 64);

  // Q fragments (already scaled by 1/sqrt(64)*log2e): A-layout, K=64
  const v16bf qa0 = load_a_bf16(Qrow, khalf);
  const v16bf qa1 = load_a_bf16(Qrow, 32 + khalf);

  v8f o0 = vzero8(), o1 = vzero8(), o2 = vzero8(), o3 = vzero8();
  float m[8], lp[8];     // running row-max and PER-LANE partial row-sum
#pragma unroll
  for (int r = 0; r < 8; ++r) { m[r] = NEG_BIG; lp[r] = 0.0f; }

  float p0[8], p1[8], p2[8], p3[8], alpha[8];

  // online-softmax step for one row group (exp2 domain; scale pre-folded)
  auto softmax_step = [&](float e0, float e1, float e2, float e3, int r) {
    float v = fmaxf(fmaxf(e0, e1), fmaxf(e2, e3));
    v = fmaxf(v, __shfl_xor(v, 1, 16));
    v = fmaxf(v, __shfl_xor(v, 2, 16));
    v = fmaxf(v, __shfl_xor(v, 4, 16));
    v = fmaxf(v, __shfl_xor(v, 8, 16));
    const float mn = fmaxf(m[r], v);
    alpha[r] = exp2f(m[r] - mn);
    m[r] = mn;
    p0[r] = exp2f(e0 - mn);
    p1[r] = exp2f(e1 - mn);
    p2[r] = exp2f(e2 - mn);
    p3[r] = exp2f(e3 - mn);
    lp[r] = lp[r] * alpha[r] + ((p0[r] + p1[r]) + (p2[r] + p3[r]));
  };

  const int nkb = blockIdx.x + 1;                // key blocks of 64 (causal)
  for (int kb = 0; kb < nkb; ++kb) {
    const int kbase = kb * 64;

    // ---- S = Q K^T : four 16x16 N-tiles, K = 64
    v8f s0 = vzero8(), s1 = vzero8(), s2 = vzero8(), s3 = vzero8();
    {
      const bf16_t* kr0 = Kbase + (size_t)(kbase + l15) * HD;
      const bf16_t* kr1 = kr0 + (size_t)16 * HD;
      const bf16_t* kr2 = kr0 + (size_t)32 * HD;
      const bf16_t* kr3 = kr0 + (size_t)48 * HD;
      s0 = wmma_bf16(qa0, *(const v16bf*)(kr0 + klo), s0);
      s0 = wmma_bf16(qa1, *(const v16bf*)(kr0 + 32 + klo), s0);
      s1 = wmma_bf16(qa0, *(const v16bf*)(kr1 + klo), s1);
      s1 = wmma_bf16(qa1, *(const v16bf*)(kr1 + 32 + klo), s1);
      s2 = wmma_bf16(qa0, *(const v16bf*)(kr2 + klo), s2);
      s2 = wmma_bf16(qa1, *(const v16bf*)(kr2 + 32 + klo), s2);
      s3 = wmma_bf16(qa0, *(const v16bf*)(kr3 + klo), s3);
      s3 = wmma_bf16(qa1, *(const v16bf*)(kr3 + 32 + klo), s3);
    }

    // ---- online softmax (wave-uniform diag/interior branch)
    if (kb == nkb - 1) {            // diagonal block: per-row causal masks
#pragma unroll
      for (int r = 0; r < 8; ++r) {
        const int q = q0 + r + h * 8;
        const float e0 = (kbase + l15      <= q) ? s0[r] : NEG_BIG;
        const float e1 = (kbase + 16 + l15 <= q) ? s1[r] : NEG_BIG;
        const float e2 = (kbase + 32 + l15 <= q) ? s2[r] : NEG_BIG;
        const float e3 = (kbase + 48 + l15 <= q) ? s3[r] : NEG_BIG;
        softmax_step(e0, e1, e2, e3, r);
      }
    } else {                        // interior block: no masking needed
#pragma unroll
      for (int r = 0; r < 8; ++r) softmax_step(s0[r], s1[r], s2[r], s3[r], r);
    }

    // ---- rescale O; stage P to LDS (D-layout -> row-major 16x64 bf16)
#pragma unroll
    for (int r = 0; r < 8; ++r) {
      o0[r] *= alpha[r]; o1[r] *= alpha[r];
      o2[r] *= alpha[r]; o3[r] *= alpha[r];
      const int row = r + h * 8;
      pw[row * 64 + l15]      = (bf16_t)p0[r];
      pw[row * 64 + 16 + l15] = (bf16_t)p1[r];
      pw[row * 64 + 32 + l15] = (bf16_t)p2[r];
      pw[row * 64 + 48 + l15] = (bf16_t)p3[r];
    }

    // ---- reload P in A-layout (per-wave LDS region, no barrier needed)
    const v16bf pa0 = load_a_bf16(pw + l15 * 64, khalf);
    const v16bf pa1 = load_a_bf16(pw + l15 * 64, 32 + khalf);

    // ---- O += P V : B column n = Vt row (16*nt + lane&15), contiguous in S
    {
      const bf16_t* vr0 = Vtb + (size_t)(l15)      * SEQ + kbase;
      const bf16_t* vr1 = Vtb + (size_t)(16 + l15) * SEQ + kbase;
      const bf16_t* vr2 = Vtb + (size_t)(32 + l15) * SEQ + kbase;
      const bf16_t* vr3 = Vtb + (size_t)(48 + l15) * SEQ + kbase;
      o0 = wmma_bf16(pa0, *(const v16bf*)(vr0 + klo), o0);
      o0 = wmma_bf16(pa1, *(const v16bf*)(vr0 + 32 + klo), o0);
      o1 = wmma_bf16(pa0, *(const v16bf*)(vr1 + klo), o1);
      o1 = wmma_bf16(pa1, *(const v16bf*)(vr1 + 32 + klo), o1);
      o2 = wmma_bf16(pa0, *(const v16bf*)(vr2 + klo), o2);
      o2 = wmma_bf16(pa1, *(const v16bf*)(vr2 + 32 + klo), o2);
      o3 = wmma_bf16(pa0, *(const v16bf*)(vr3 + klo), o3);
      o3 = wmma_bf16(pa1, *(const v16bf*)(vr3 + 32 + klo), o3);
    }
  }

  // epilogue: cross-lane reduce lp once, normalize, store out[:, 128:192]
#pragma unroll
  for (int r = 0; r < 8; ++r) {
    float l = lp[r];
    l += __shfl_xor(l, 1, 16);
    l += __shfl_xor(l, 2, 16);
    l += __shfl_xor(l, 4, 16);
    l += __shfl_xor(l, 8, 16);
    const float inv = 1.0f / l;
    float* orow = out + (size_t)(b * SEQ + q0 + r + h * 8) * OUTW + DIM;
    orow[l15]      = o0[r] * inv;
    orow[16 + l15] = o1[r] * inv;
    orow[32 + l15] = o2[r] * inv;
    orow[48 + l15] = o3[r] * inv;
  }
}

// ---------------------------------------------------------------- launch

extern "C" void kernel_launch(void* const* d_in, const int* in_sizes, int n_in,
                              void* d_out, int out_size, void* d_ws, size_t ws_size,
                              hipStream_t stream) {
  const float* x  = (const float*)d_in[0];
  const float* Wq = (const float*)d_in[1];
  const float* bq = (const float*)d_in[2];
  const float* Wk = (const float*)d_in[3];
  const float* bk = (const float*)d_in[4];
  const float* Wv = (const float*)d_in[5];
  const float* bv = (const float*)d_in[6];
  float* out = (float*)d_out;

  char* ws = (char*)d_ws;
  bf16_t* Wt = (bf16_t*)ws;                                  // 48 KB
  bf16_t* Qb = (bf16_t*)(ws + (64 << 10));                   // 2 MB
  bf16_t* Kb = (bf16_t*)(ws + (64 << 10) + (2 << 20));       // 2 MB
  bf16_t* Vt = (bf16_t*)(ws + (64 << 10) + (4 << 20));       // 2 MB

  prep_weights_kernel<<<(3 * HD * DIM) / 256, 256, 0, stream>>>(Wq, Wk, Wv, Wt);
  qkv_proj_kernel<<<(BATCH * SEQ) / 16, 128, 0, stream>>>(x, Wt, bq, bk, bv, Qb, Kb, Vt);
  copy_inputs_kernel<<<(BATCH * SEQ * 32) / 256, 256, 0, stream>>>((const float4*)x, out);
  attn_kernel<<<dim3(SEQ / 64, BATCH), 128, 0, stream>>>(Qb, Kb, Vt, out);
}
